// SelfAttention_57913339019860
// MI455X (gfx1250) — matrix operand order repaired and checked by reading, not compiled, not run on hardware
//
#include <hip/hip_runtime.h>

typedef __attribute__((ext_vector_type(16))) __bf16 v16bf;
typedef __attribute__((ext_vector_type(8)))  __bf16 v8bf;
typedef __attribute__((ext_vector_type(2)))  __bf16 bf2;
typedef __attribute__((ext_vector_type(8)))  float  v8f;
typedef __attribute__((ext_vector_type(8)))  unsigned int v8u;

// Problem constants (from reference): B=2, L=4096, H=8, E=D=64
#define BATCH 2
#define SEQ   4096
#define HEADS 8
#define EDIM  64

#define NTHR 128 // 4 waves
#define BQ   128 // queries per workgroup (4 waves x 32)
#define BK   64  // keys/values per KV iteration (36 WMMA per barrier pair)
#define KP   72  // Klds row stride in bf16 (64 + 8 pad, keeps 16B alignment)
#define VP   72  // Vtlds row stride in bf16 (64 keys + 8 pad, even -> b32-aligned pairs)

#if __has_builtin(__builtin_amdgcn_exp2f)
#define EXP2(x) __builtin_amdgcn_exp2f(x)     // raw v_exp_f32, no denorm guard
#else
#define EXP2(x) __builtin_exp2f(x)
#endif
#if __has_builtin(__builtin_amdgcn_rcpf)
#define RCP(x) __builtin_amdgcn_rcpf(x)
#else
#define RCP(x) (1.0f / (x))
#endif

static __device__ __forceinline__ unsigned int pack_bf16(float a, float b) {
    bf2 p = { (__bf16)a, (__bf16)b };          // -> v_cvt_pk_bf16_f32
    return __builtin_bit_cast(unsigned int, p);
}

__global__ __launch_bounds__(NTHR)
void fattn_wmma_bf16(const float* __restrict__ Q,
                     const float* __restrict__ K,
                     const float* __restrict__ V,
                     float* __restrict__ O)
{
    __shared__ __bf16 Klds[BK * KP];    // [key][e]   bf16
    __shared__ __bf16 Vtlds[EDIM * VP]; // [d][key]   bf16 (V transposed)

    const int h  = blockIdx.y;
    const int b  = blockIdx.z;
    const int q0 = blockIdx.x * BQ;

    const int tid    = threadIdx.x;
    const int lane   = tid & 31;
    const int wave   = tid >> 5;        // 0..3, each owns 32 queries
    const int lane16 = lane & 15;
    const int half   = lane >> 4;       // 0 = lanes 0-15, 1 = lanes 16-31

    const long strideL = (long)HEADS * EDIM;
    const long headOff = (((long)b * SEQ) * HEADS + h) * EDIM;
    const float* Qb = Q + headOff;
    const float* Kb = K + headOff;
    const float* Vb = V + headOff;
    float*       Ob = O + headOff;

    // softmax scale folded into Q, exp2 domain: (1/sqrt(64)) * log2(e)
    const float sc = 0.125f * 1.44269504088896340f;

    // ---- constant ones A-fragment: A-layout row 0 lives entirely in lanes with
    //      lane16==0 (lane 0: K=0..15; lane 16: K=16..31).
    //      A x B with this A yields row 0 = column sums of B (= row-sum of P).
    v16bf aone;
    {
        const __bf16 one = (__bf16)(lane16 == 0 ? 1.0f : 0.0f);
        #pragma unroll
        for (int j = 0; j < 16; ++j) aone[j] = one;
    }

    // ---- Q fragments (B-layout of Q^T): k(=e) = 32c + half*16 + j, n(=q) = lane16
    int qrow[2];
    v16bf qfrag[2][2];
    #pragma unroll
    for (int qt = 0; qt < 2; ++qt) {
        qrow[qt] = q0 + wave * 32 + qt * 16 + lane16;
        const float* qp = Qb + (long)qrow[qt] * strideL;
        #pragma unroll
        for (int c = 0; c < 2; ++c) {
            const int e0 = 32 * c + half * 16;
            #pragma unroll
            for (int j = 0; j < 16; ++j)
                qfrag[qt][c][j] = (__bf16)(qp[e0 + j] * sc);
        }
    }

    // per q-tile: acc[0..3] = O^T d-tiles (16 d x 16 q), acc[4] row0 = l (row-sum)
    v8f acc[2][5];
    #pragma unroll
    for (int qt = 0; qt < 2; ++qt)
        #pragma unroll
        for (int dt = 0; dt < 5; ++dt)
            acc[qt][dt] = (v8f){0.f,0.f,0.f,0.f,0.f,0.f,0.f,0.f};

    float mrow[2] = { -1e30f, -1e30f };

    for (int kv = 0; kv < SEQ; kv += BK) {
        // ---- stage 2 keys x 16 dims per thread:
        //      K -> Klds[key][e] (pairs packed along e)
        //      V -> Vtlds[d][key] (pairs packed along key: one b32 per d)
        {
            const int s0 = (tid >> 2) * 2;    // even key 0..62
            const int d0 = (tid & 3) * 16;    // 0,16,32,48
            const float* kp = Kb + (long)(kv + s0) * strideL + d0;
            const float* vp = Vb + (long)(kv + s0) * strideL + d0;

            float k0[16], k1[16], v0[16], v1[16];
            #pragma unroll
            for (int j = 0; j < 16; ++j) {
                k0[j] = kp[j];            k1[j] = kp[strideL + j];
                v0[j] = vp[j];            v1[j] = vp[strideL + j];
            }
            #pragma unroll
            for (int r = 0; r < 2; ++r) {
                const float* kr = r ? k1 : k0;
                v8bf lo, hi;
                #pragma unroll
                for (int j = 0; j < 8; ++j) { lo[j] = (__bf16)kr[j]; hi[j] = (__bf16)kr[8 + j]; }
                *(v8bf*)&Klds[(s0 + r) * KP + d0]     = lo;   // 16B aligned
                *(v8bf*)&Klds[(s0 + r) * KP + d0 + 8] = hi;
            }
            #pragma unroll
            for (int j = 0; j < 16; ++j)      // key-pair pack -> single b32 store
                *(unsigned int*)&Vtlds[(d0 + j) * VP + s0] = pack_bf16(v0[j], v1[j]);

            if (kv + BK < SEQ) {              // prefetch next tile (global_prefetch_b8)
                __builtin_prefetch(kp + (long)BK * strideL, 0, 3);
                __builtin_prefetch(vp + (long)BK * strideL, 0, 3);
            }
        }
        __syncthreads();

        #pragma unroll
        for (int qt = 0; qt < 2; ++qt) {
            // ---- S^T tiles: st[kt] = (keys 16kt..16kt+15) x (16 queries)
            v8f st[4];
            #pragma unroll
            for (int kt = 0; kt < 4; ++kt) {
                v8f c = (v8f){0.f,0.f,0.f,0.f,0.f,0.f,0.f,0.f};
                #pragma unroll
                for (int cc = 0; cc < 2; ++cc) {
                    // A-layout: row(key)=lane16, e runs base+0..15 (base=32cc+half*16)
                    v16bf a;
                    const __bf16* kr = &Klds[(kt * 16 + lane16) * KP + 32 * cc + half * 16];
                    #pragma unroll
                    for (int j = 0; j < 8; ++j) a[j]     = kr[j];
                    #pragma unroll
                    for (int j = 0; j < 8; ++j) a[8 + j] = kr[8 + j];
                    c = __builtin_amdgcn_wmma_f32_16x16x32_bf16(
                            false, a, false, qfrag[qt][cc], (short)0, c, false, false);
                }
                st[kt] = c;
            }

            // ---- online softmax max (in-lane rows + one xor16 shuffle)
            float mx = -1e30f;
            #pragma unroll
            for (int kt = 0; kt < 4; ++kt)
                #pragma unroll
                for (int i = 0; i < 8; ++i) mx = fmaxf(mx, st[kt][i]);
            mx = fmaxf(mx, __shfl_xor(mx, 16, 32));
            const float mnew = fmaxf(mrow[qt], mx);
            const float alpha = EXP2(mrow[qt] - mnew);
            mrow[qt] = mnew;

            #pragma unroll
            for (int kt = 0; kt < 4; ++kt)
                #pragma unroll
                for (int i = 0; i < 8; ++i)
                    st[kt][i] = EXP2(st[kt][i] - mnew);

            // ---- P^T C-layout -> B-layout per 32-key chunk, on packed bf16 dwords
            v16bf bp[2];
            #pragma unroll
            for (int cc = 0; cc < 2; ++cc) {
                unsigned int pka[4], pkb[4];
                #pragma unroll
                for (int i = 0; i < 4; ++i) {
                    pka[i] = pack_bf16(st[2*cc][2*i],     st[2*cc][2*i + 1]);
                    pkb[i] = pack_bf16(st[2*cc + 1][2*i], st[2*cc + 1][2*i + 1]);
                }
                v8u bu;
                #pragma unroll
                for (int i = 0; i < 4; ++i) {
                    const unsigned int xa = (unsigned int)__shfl_xor((int)pka[i], 16, 32);
                    const unsigned int xb = (unsigned int)__shfl_xor((int)pkb[i], 16, 32);
                    // low : keys 0..7 own kt_a, keys 8..15 partner kt_a
                    // high: keys 16..23 partner kt_b, keys 24..31 own kt_b
                    bu[i]     = half ? xb     : pka[i];
                    bu[4 + i] = half ? pkb[i] : xa;
                }
                bp[cc] = __builtin_bit_cast(v16bf, bu);
            }

            // ---- O^T(d,q) = alpha*O^T + V^T x P^T   (4 d-tiles x 2 key-chunks)
            #pragma unroll
            for (int dt = 0; dt < 4; ++dt) {
                v8f c = acc[qt][dt];
                #pragma unroll
                for (int i = 0; i < 8; ++i) c[i] *= alpha;
                #pragma unroll
                for (int cc = 0; cc < 2; ++cc) {
                    v16bf av;  // A-layout: row(d)=dt*16+lane16, key runs 32cc+half*16 (+0..15)
                    const __bf16* vr = &Vtlds[(dt * 16 + lane16) * VP + 32 * cc + half * 16];
                    #pragma unroll
                    for (int j = 0; j < 8; ++j) av[j]     = vr[j];
                    #pragma unroll
                    for (int j = 0; j < 8; ++j) av[8 + j] = vr[8 + j];
                    c = __builtin_amdgcn_wmma_f32_16x16x32_bf16(
                            false, av, false, bp[cc], (short)0, c, false, false);
                }
                acc[qt][dt] = c;
            }

            // ---- l(q) via matrix pipe: row 0 of (ones x P^T), same alpha recurrence
            {
                v8f c = acc[qt][4];
                #pragma unroll
                for (int i = 0; i < 8; ++i) c[i] *= alpha;
                c = __builtin_amdgcn_wmma_f32_16x16x32_bf16(
                        false, aone, false, bp[0], (short)0, c, false, false);
                c = __builtin_amdgcn_wmma_f32_16x16x32_bf16(
                        false, aone, false, bp[1], (short)0, c, false, false);
                acc[qt][4] = c;
            }
        }
        __syncthreads();
    }

    // ---- epilogue: out[q][d] = acc / l   (8 contiguous floats per dt per lane)
    #pragma unroll
    for (int qt = 0; qt < 2; ++qt) {
        // l lives in C row 0 => vgpr 0 of low-half lanes; broadcast to high half
        const float l0 = acc[qt][4][0];
        const float lx = __shfl_xor(l0, 16, 32);
        const float inv = RCP(half ? lx : l0);
        float* op = Ob + (long)qrow[qt] * strideL;
        #pragma unroll
        for (int dt = 0; dt < 4; ++dt) {
            const int d0 = dt * 16 + half * 8;   // C rows: M = i + half*8 within d-tile
            #pragma unroll
            for (int i = 0; i < 8; ++i)
                op[d0 + i] = acc[qt][dt][i] * inv;
        }
    }
}

extern "C" void kernel_launch(void* const* d_in, const int* in_sizes, int n_in,
                              void* d_out, int out_size, void* d_ws, size_t ws_size,
                              hipStream_t stream) {
    (void)in_sizes; (void)n_in; (void)out_size; (void)d_ws; (void)ws_size;
    const float* Q = (const float*)d_in[0];
    const float* K = (const float*)d_in[1];
    const float* V = (const float*)d_in[2];
    float* O = (float*)d_out;

    dim3 grid(SEQ / BQ, HEADS, BATCH);   // (32, 8, 2)
    dim3 block(NTHR);
    fattn_wmma_bf16<<<grid, block, 0, stream>>>(Q, K, V, O);
}
